// BiDAF_25890062860730
// MI455X (gfx1250) — compile-verified, matrix operand order checked by reading, not run on hardware
//
#include <hip/hip_runtime.h>
#include <hip/hip_bf16.h>

// ---------------------------------------------------------------------------
// BiDAF / QANet forward on gfx1250 (MI455X).
// All matmuls run through one batched WMMA GEMM kernel:
//   - 4 waves / block, each wave owns a 16x64 C tile (4 accumulators)
//   - branch-free main K loop (M/N handled by clamping), float2/v8h loads
//   - weight matrices pre-converted to f16 once per launch (B fp16 fast path:
//     two 16B loads per fragment, no per-use cvt)
//   - single guarded zero-filled K remainder step
// ---------------------------------------------------------------------------

typedef __attribute__((ext_vector_type(16))) _Float16 v16h;
typedef __attribute__((ext_vector_type(8)))  _Float16 v8h;
typedef __attribute__((ext_vector_type(8)))  float    v8f;

#define H_DIM 128
#define FF_DIM 512
#define NEGV (-1e30f)

// ISA 7.12.2: 16-bit A 16x32 fragment K-pair base for vector-reg j, lane-half h
__device__ __forceinline__ int frag_kb(int j, int half) {
    return (j < 4) ? (half * 8 + 2 * j) : (16 + half * 8 + 2 * (j - 4));
}

__device__ __forceinline__ v16h join8(v8h lo, v8h hi) {
    return __builtin_shufflevector(lo, hi, 0, 1, 2, 3, 4, 5, 6, 7,
                                           8, 9, 10, 11, 12, 13, 14, 15);
}

// ======================= WMMA GEMM =========================================
// C[b] = act(A[b] @ op(B[b]) + bias) + residual
//   A: M x K row-major f32 (lda)
//   op(B): K x N.  bHalf==1: B is f16, N x K row-major (weights, transB).
//                  bHalf==0: transB selects N x K / K x N row-major f32.
//   act: 0 none, 1 relu, 2 sigmoid
__global__ __launch_bounds__(128)
void gemm_wmma_kernel(const float* __restrict__ A, const void* __restrict__ Bv,
                      const float* __restrict__ bias, const float* __restrict__ res,
                      float* __restrict__ C,
                      int M, int N, int K, int lda, int ldb, int ldc,
                      long long sA, long long sB, long long sC, long long sR,
                      int transB, int bHalf, int act)
{
    const int lane = threadIdx.x & 31;
    const int wave = threadIdx.x >> 5;
    const int half = lane >> 4;
    const int l16  = lane & 15;
    const int tM = (blockIdx.y * 4 + wave) * 16;
    const int tN = blockIdx.x * 64;
    const int b  = blockIdx.z;

    const float*    Ab = A + (long long)b * sA;
    const float*    Bf = (const float*)Bv + (long long)b * sB;
    const _Float16* Bh = (const _Float16*)Bv;   // weights are never batched
    float*          Cb = C + (long long)b * sC;
    const float*    Rb = res ? (res + (long long)b * sR) : nullptr;

    // Clamped fragment coordinates: OOB rows/cols compute garbage that is
    // never stored (A row m only affects C row m; B col n only C col n).
    const int am  = tM + l16;
    const int amc = am < M ? am : (M - 1);
    int bnc[4];
#pragma unroll
    for (int t = 0; t < 4; ++t) {
        int bn = tN + t * 16 + l16;
        bnc[t] = bn < N ? bn : (N - 1);
    }

    const float* Arow = Ab + (long long)amc * lda;

    v8f acc0 = {}, acc1 = {}, acc2 = {}, acc3 = {};
    const int K32 = K & ~31;

    for (int k0 = 0; k0 < K32; k0 += 32) {
        if (k0 + 32 < K32)  // wave-uniform -> plain scalar branch
            __builtin_prefetch(&Arow[k0 + 32], 0, 1);
        v16h af;
#pragma unroll
        for (int j = 0; j < 8; ++j) {
            const float2 a2 = *(const float2*)&Arow[k0 + frag_kb(j, half)];
            af[2 * j]     = (_Float16)a2.x;
            af[2 * j + 1] = (_Float16)a2.y;
        }
        v16h bf0, bf1, bf2, bf3;
        if (bHalf) {
            // fragment elems [0..7] = K half*8+0..7 ; [8..15] = K 16+half*8+0..7
            const long long kb0 = k0 + half * 8;
            bf0 = join8(*(const v8h*)&Bh[(long long)bnc[0] * ldb + kb0],
                        *(const v8h*)&Bh[(long long)bnc[0] * ldb + kb0 + 16]);
            bf1 = join8(*(const v8h*)&Bh[(long long)bnc[1] * ldb + kb0],
                        *(const v8h*)&Bh[(long long)bnc[1] * ldb + kb0 + 16]);
            bf2 = join8(*(const v8h*)&Bh[(long long)bnc[2] * ldb + kb0],
                        *(const v8h*)&Bh[(long long)bnc[2] * ldb + kb0 + 16]);
            bf3 = join8(*(const v8h*)&Bh[(long long)bnc[3] * ldb + kb0],
                        *(const v8h*)&Bh[(long long)bnc[3] * ldb + kb0 + 16]);
        } else if (transB) {
#pragma unroll
            for (int j = 0; j < 8; ++j) {
                const int kb = k0 + frag_kb(j, half);
                float2 b0 = *(const float2*)&Bf[(long long)bnc[0] * ldb + kb];
                float2 b1 = *(const float2*)&Bf[(long long)bnc[1] * ldb + kb];
                float2 b2 = *(const float2*)&Bf[(long long)bnc[2] * ldb + kb];
                float2 b3 = *(const float2*)&Bf[(long long)bnc[3] * ldb + kb];
                bf0[2 * j] = (_Float16)b0.x; bf0[2 * j + 1] = (_Float16)b0.y;
                bf1[2 * j] = (_Float16)b1.x; bf1[2 * j + 1] = (_Float16)b1.y;
                bf2[2 * j] = (_Float16)b2.x; bf2[2 * j + 1] = (_Float16)b2.y;
                bf3[2 * j] = (_Float16)b3.x; bf3[2 * j + 1] = (_Float16)b3.y;
            }
        } else {
#pragma unroll
            for (int j = 0; j < 8; ++j) {
                const int kb = k0 + frag_kb(j, half);
                const float* r0 = Bf + (long long)kb * ldb;
                const float* r1 = Bf + (long long)(kb + 1) * ldb;
                bf0[2 * j] = (_Float16)r0[bnc[0]]; bf0[2 * j + 1] = (_Float16)r1[bnc[0]];
                bf1[2 * j] = (_Float16)r0[bnc[1]]; bf1[2 * j + 1] = (_Float16)r1[bnc[1]];
                bf2[2 * j] = (_Float16)r0[bnc[2]]; bf2[2 * j + 1] = (_Float16)r1[bnc[2]];
                bf3[2 * j] = (_Float16)r0[bnc[3]]; bf3[2 * j + 1] = (_Float16)r1[bnc[3]];
            }
        }
        acc0 = __builtin_amdgcn_wmma_f32_16x16x32_f16(false, af, false, bf0, (short)0, acc0, false, false);
        acc1 = __builtin_amdgcn_wmma_f32_16x16x32_f16(false, af, false, bf1, (short)0, acc1, false, false);
        acc2 = __builtin_amdgcn_wmma_f32_16x16x32_f16(false, af, false, bf2, (short)0, acc2, false, false);
        acc3 = __builtin_amdgcn_wmma_f32_16x16x32_f16(false, af, false, bf3, (short)0, acc3, false, false);
    }

    if (K32 < K) {  // single zero-filled remainder step (K mixes into all outputs)
        v16h af;
        v16h bf[4];
#pragma unroll
        for (int j = 0; j < 8; ++j) {
            const int kb = frag_kb(j, half);
            const int k  = K32 + kb;
            af[2 * j]     = (k     < K) ? (_Float16)Arow[k]     : (_Float16)0.f;
            af[2 * j + 1] = (k + 1 < K) ? (_Float16)Arow[k + 1] : (_Float16)0.f;
#pragma unroll
            for (int t = 0; t < 4; ++t) {
                _Float16 b0 = (_Float16)0.f, b1 = (_Float16)0.f;
                if (bHalf) {
                    if (k     < K) b0 = Bh[(long long)bnc[t] * ldb + k];
                    if (k + 1 < K) b1 = Bh[(long long)bnc[t] * ldb + k + 1];
                } else if (transB) {
                    if (k     < K) b0 = (_Float16)Bf[(long long)bnc[t] * ldb + k];
                    if (k + 1 < K) b1 = (_Float16)Bf[(long long)bnc[t] * ldb + k + 1];
                } else {
                    if (k     < K) b0 = (_Float16)Bf[(long long)k * ldb + bnc[t]];
                    if (k + 1 < K) b1 = (_Float16)Bf[(long long)(k + 1) * ldb + bnc[t]];
                }
                bf[t][2 * j] = b0; bf[t][2 * j + 1] = b1;
            }
        }
        acc0 = __builtin_amdgcn_wmma_f32_16x16x32_f16(false, af, false, bf[0], (short)0, acc0, false, false);
        acc1 = __builtin_amdgcn_wmma_f32_16x16x32_f16(false, af, false, bf[1], (short)0, acc1, false, false);
        acc2 = __builtin_amdgcn_wmma_f32_16x16x32_f16(false, af, false, bf[2], (short)0, acc2, false, false);
        acc3 = __builtin_amdgcn_wmma_f32_16x16x32_f16(false, af, false, bf[3], (short)0, acc3, false, false);
    }

    const v8f accs[4] = { acc0, acc1, acc2, acc3 };
#pragma unroll
    for (int t = 0; t < 4; ++t) {
        const int n = tN + t * 16 + l16;
        if (n >= N) continue;
#pragma unroll
        for (int i = 0; i < 8; ++i) {
            const int m = tM + i + 8 * half;
            if (m < M) {
                float v = accs[t][i];
                if (bias) v += bias[n];
                if (act == 1)      v = v > 0.f ? v : 0.f;
                else if (act == 2) v = 1.f / (1.f + __expf(-v));
                if (Rb) v += Rb[(long long)m * ldc + n];
                Cb[(long long)m * ldc + n] = v;
            }
        }
    }
}

// f32-B (activation operand) GEMM
static inline void gemm(hipStream_t st, const float* A, const float* B,
                        const float* bias, const float* res, float* C,
                        int M, int N, int K, int lda, int ldb, int ldc,
                        long long sA, long long sB, long long sC, long long sR,
                        int batch, int transB, int act)
{
    dim3 g((N + 63) / 64, (M + 63) / 64, batch);
    gemm_wmma_kernel<<<g, 128, 0, st>>>(A, B, bias, res, C, M, N, K, lda, ldb, ldc,
                                        sA, sB, sC, sR, transB, 0, act);
}

// f16-B (pre-converted weight, x @ W^T) GEMM
static inline void gemmW(hipStream_t st, const float* A, const _Float16* Wh,
                         const float* bias, const float* res, float* C,
                         int M, int N, int K, int ldc, int act)
{
    dim3 g((N + 63) / 64, (M + 63) / 64, 1);
    gemm_wmma_kernel<<<g, 128, 0, st>>>(A, Wh, bias, res, C, M, N, K, K, K, ldc,
                                        0, 0, 0, 0, 1, 1, act);
}

// ======================= support kernels ===================================

__global__ void f16cvt_kernel(const float* __restrict__ s, _Float16* __restrict__ d,
                              long long n)
{
    long long i = blockIdx.x * (long long)blockDim.x + threadIdx.x;
    if (i < n) d[i] = (_Float16)s[i];
}

__device__ __forceinline__ float wave_sum(float v) {
    for (int o = 16; o > 0; o >>= 1) v += __shfl_xor(v, o, 32);
    return v;
}
__device__ __forceinline__ float wave_max(float v) {
    for (int o = 16; o > 0; o >>= 1) v = fmaxf(v, __shfl_xor(v, o, 32));
    return v;
}

__global__ __launch_bounds__(32)
void layernorm_kernel(const float* __restrict__ X, float* __restrict__ Y,
                      const float* __restrict__ g, const float* __restrict__ bta, int H)
{
    const long long row = blockIdx.x;
    const int lane = threadIdx.x;
    const float* x = X + row * H;
    float s = 0.f;
    for (int j = lane; j < H; j += 32) s += x[j];
    s = wave_sum(s);
    const float mean = s / (float)H;
    float v = 0.f;
    for (int j = lane; j < H; j += 32) { float d = x[j] - mean; v += d * d; }
    v = wave_sum(v);
    const float rstd = rsqrtf(v / (float)H + 1e-5f);
    float* y = Y + row * H;
    for (int j = lane; j < H; j += 32) y[j] = (x[j] - mean) * rstd * g[j] + bta[j];
}

__global__ __launch_bounds__(32)
void softmax_kernel(const float* __restrict__ Sin, float* __restrict__ Sout,
                    const float* __restrict__ maskf, int N,
                    long long batchStride, long long rowStride, int elemStride,
                    long long maskBatchStride, float scale)
{
    const int r = blockIdx.x, b = blockIdx.y, lane = threadIdx.x;
    const long long base = (long long)b * batchStride + (long long)r * rowStride;
    const float* mrow = maskf + (long long)b * maskBatchStride;
    float mx = -3.4e38f;
    for (int j = lane; j < N; j += 32)
        mx = fmaxf(mx, Sin[base + (long long)j * elemStride] * scale + mrow[j] * NEGV);
    mx = wave_max(mx);
    float sum = 0.f;
    for (int j = lane; j < N; j += 32)
        sum += __expf(Sin[base + (long long)j * elemStride] * scale + mrow[j] * NEGV - mx);
    sum = wave_sum(sum);
    const float inv = 1.f / sum;
    for (int j = lane; j < N; j += 32) {
        float v = Sin[base + (long long)j * elemStride] * scale + mrow[j] * NEGV;
        Sout[base + (long long)j * elemStride] = __expf(v - mx) * inv;
    }
}

__global__ void posenc_kernel(float* __restrict__ X, int L, int H)
{
    const int r = blockIdx.x, h = threadIdx.x;
    const int l = r % L;
    const int num_ts = L / 2;
    const float denom = (float)(num_ts - 1);
    float add = 0.f;
    if (l < num_ts) {
        float inv = __expf((float)l * (-9.210340371976184f / denom));
        add = __sinf((float)h * inv);
    } else if (l < 2 * num_ts) {
        float inv = __expf((float)(l - num_ts) * (-9.210340371976184f / denom));
        add = __cosf((float)h * inv);
    }
    X[(long long)r * H + h] += add;
}

__global__ void dwconv_kernel(const float* __restrict__ X, float* __restrict__ Y,
                              const float* __restrict__ w, const float* __restrict__ bc,
                              int L, int H, int Kk)
{
    const int l = blockIdx.x, b = blockIdx.y, c = threadIdx.x;
    const int pad = Kk / 2;
    const float* xb = X + (long long)b * L * H;
    float acc = bc[c];
    for (int t = 0; t < Kk; ++t) {
        int ll = l + t - pad;
        if (ll >= 0 && ll < L) acc += xb[(long long)ll * H + c] * w[c * Kk + t];
    }
    Y[((long long)b * L + l) * H + c] = acc;
}

__global__ __launch_bounds__(128)
void charconv_kernel(const int* __restrict__ ids, const float* __restrict__ emb,
                     const float* __restrict__ w, const float* __restrict__ bc,
                     float* __restrict__ out, int ldo)
{
    __shared__ float e[16][8];
    const int r = blockIdx.x, t = threadIdx.x;
    { int p = t >> 3, d = t & 7; e[p][d] = emb[ids[r * 16 + p] * 8 + d]; }
    __syncthreads();
    if (t < 100) {
        float best = -3.4e38f;
        for (int p = 0; p < 12; ++p) {
            float acc = bc[t];
            for (int d = 0; d < 8; ++d)
#pragma unroll
                for (int kk = 0; kk < 5; ++kk)
                    acc += e[p + kk][d] * w[(t * 8 + d) * 5 + kk];
            best = fmaxf(best, acc);
        }
        out[(long long)r * ldo + t] = best;
    }
}

__global__ void wordgather_kernel(const int* __restrict__ wid, const float* __restrict__ wemb,
                                  float* __restrict__ out, int ldo)
{
    const int r = blockIdx.x, j = threadIdx.x;
    if (j < 100) out[(long long)r * ldo + 100 + j] = wemb[(long long)wid[r] * 100 + j];
}

__global__ void mask_kernel(const int* __restrict__ wid, float* __restrict__ m, int n)
{
    int i = blockIdx.x * blockDim.x + threadIdx.x;
    if (i < n) m[i] = (wid[i] == 1) ? 1.f : 0.f;
}

__global__ void highway_combine_kernel(float* __restrict__ x, const float* __restrict__ h,
                                       const float* __restrict__ g, long long n)
{
    long long i = blockIdx.x * (long long)blockDim.x + threadIdx.x;
    if (i < n) { float gg = g[i]; x[i] = gg * h[i] + (1.f - gg) * x[i]; }
}

__global__ void colscale_kernel(const float* __restrict__ x, const float* __restrict__ v,
                                float* __restrict__ y, int H)
{
    long long r = blockIdx.x; int h = threadIdx.x;
    y[r * H + h] = x[r * H + h] * v[h];
}

__global__ __launch_bounds__(32)
void rowdot_kernel(const float* __restrict__ x, const float* __restrict__ v,
                   float* __restrict__ out, int H)
{
    long long r = blockIdx.x; int lane = threadIdx.x;
    float s = 0.f;
    for (int j = lane; j < H; j += 32) s += x[r * H + j] * v[j];
    s = wave_sum(s);
    if (lane == 0) out[r] = s;
}

__global__ void builds_kernel(float* __restrict__ s, const float* __restrict__ cA,
                              const float* __restrict__ qA, const float* __restrict__ biasp,
                              int LC, int LQ)
{
    int c = blockIdx.x, b = blockIdx.y, q = threadIdx.x;
    if (q < LQ)
        s[((long long)b * LC + c) * LQ + q] += cA[b * LC + c] + qA[b * LQ + q] + biasp[0];
}

__global__ void gconcat_kernel(const float* __restrict__ c, const float* __restrict__ c2q,
                               const float* __restrict__ q2c, float* __restrict__ g)
{
    long long r = blockIdx.x; int h = threadIdx.x;
    float cv = c[r * 128 + h], a = c2q[r * 128 + h], bq = q2c[r * 128 + h];
    float* gr = g + r * 512;
    gr[h] = cv; gr[128 + h] = a; gr[256 + h] = cv * a; gr[384 + h] = cv * bq;
}

__global__ __launch_bounds__(32)
void pout_kernel(const float* __restrict__ Ma, const float* __restrict__ Mb,
                 const float* __restrict__ w, const float* __restrict__ maskf,
                 float* __restrict__ out)
{
    long long r = blockIdx.x; int lane = threadIdx.x;
    float s = 0.f;
    for (int j = lane; j < 128; j += 32)
        s += Ma[r * 128 + j] * w[j] + Mb[r * 128 + j] * w[128 + j];
    s = wave_sum(s);
    if (lane == 0) out[r] = s + NEGV * maskf[r];
}

// ======================= host orchestration ================================

namespace {

struct BlockParams {
    const float *dw, *dwb, *pwb, *ln_g, *ln_b;
    const _Float16 *pw;                 // f16, n_convs x (H x H)
    int n_convs, k;
    const _Float16 *in_w, *out_w, *ff1_w, *ff2_w;   // f16 weights
    const float *in_b, *out_b, *ff1_b, *ff2_b;
    const float *n1_g, *n1_b, *n2_g, *n2_b;
};

void encoder_block(hipStream_t st, float* X, int Bb, int L, const float* maskf,
                   const BlockParams& p, float* tmp1, float* tmp2, float* scores)
{
    const int H = H_DIM, FF = FF_DIM;
    const int rows = Bb * L;
    posenc_kernel<<<rows, H, 0, st>>>(X, L, H);
    for (int i = 0; i < p.n_convs; ++i) {
        layernorm_kernel<<<rows, 32, 0, st>>>(X, tmp1, p.ln_g + i * H, p.ln_b + i * H, H);
        dwconv_kernel<<<dim3(L, Bb), H, 0, st>>>(tmp1, tmp2, p.dw + i * H * p.k,
                                                 p.dwb + i * H, L, H, p.k);
        gemmW(st, tmp2, p.pw + (long long)i * H * H, p.pwb + i * H, X, X, rows, H, H, H, 1);
    }
    gemmW(st, X, p.in_w, p.in_b, nullptr, tmp1, rows, 3 * H, H, 3 * H, 0);
    gemm(st, tmp1, tmp1 + H, nullptr, nullptr, scores, L, L, H, 3 * H, 3 * H, L,
         (long long)L * 3 * H, (long long)L * 3 * H, (long long)L * L, 0, Bb, 1, 0);
    softmax_kernel<<<dim3(L, Bb), 32, 0, st>>>(scores, scores, maskf, L,
         (long long)L * L, (long long)L, 1, (long long)L, 0.08838834764831845f);
    gemm(st, scores, tmp1 + 2 * H, nullptr, nullptr, tmp2, L, H, L, L, 3 * H, H,
         (long long)L * L, (long long)L * 3 * H, (long long)L * H, 0, Bb, 0, 0);
    gemmW(st, tmp2, p.out_w, p.out_b, X, tmp1, rows, H, H, H, 0);
    layernorm_kernel<<<rows, 32, 0, st>>>(tmp1, X, p.n1_g, p.n1_b, H);
    gemmW(st, X, p.ff1_w, p.ff1_b, nullptr, tmp2, rows, FF, H, FF, 1);
    gemmW(st, tmp2, p.ff2_w, p.ff2_b, X, tmp1, rows, H, FF, H, 0);
    layernorm_kernel<<<rows, 32, 0, st>>>(tmp1, X, p.n2_g, p.n2_b, H);
}

inline _Float16* cvtW(hipStream_t st, const float* src, _Float16*& arena, long long n)
{
    _Float16* dst = arena; arena += n;
    f16cvt_kernel<<<(int)((n + 255) / 256), 256, 0, st>>>(src, dst, n);
    return dst;
}

} // namespace

extern "C" void kernel_launch(void* const* d_in, const int* in_sizes, int n_in,
                              void* d_out, int out_size, void* d_ws, size_t ws_size,
                              hipStream_t stream)
{
    (void)in_sizes; (void)n_in; (void)out_size; (void)ws_size;
    const int B = 32, LC = 400, LQ = 50, H = H_DIM;
    const int BLc = B * LC, BLq = B * LQ;

    // inputs: dict order; params flattened alphabetically (jax tree order)
    const int*   c_char = (const int*)d_in[0];
    const int*   q_char = (const int*)d_in[1];
    const int*   c_word = (const int*)d_in[2];
    const int*   q_word = (const int*)d_in[3];
    const float* att_bias   = (const float*)d_in[4];
    const float* att_c      = (const float*)d_in[5];
    const float* att_cq     = (const float*)d_in[6];
    const float* att_q      = (const float*)d_in[7];
    const float* c_resize_w = (const float*)d_in[8];
    const float* char_conv_b= (const float*)d_in[9];
    const float* char_conv_w= (const float*)d_in[10];
    const float* char_emb   = (const float*)d_in[11];
    const float* eb_dw   = (const float*)d_in[12];
    const float* eb_dwb  = (const float*)d_in[13];
    const float* eb_ff1b = (const float*)d_in[14];
    const float* eb_ff1w = (const float*)d_in[15];
    const float* eb_ff2b = (const float*)d_in[16];
    const float* eb_ff2w = (const float*)d_in[17];
    const float* eb_inb  = (const float*)d_in[18];
    const float* eb_inw  = (const float*)d_in[19];
    const float* eb_lnb  = (const float*)d_in[20];
    const float* eb_lng  = (const float*)d_in[21];
    const float* eb_n1b  = (const float*)d_in[22];
    const float* eb_n1g  = (const float*)d_in[23];
    const float* eb_n2b  = (const float*)d_in[24];
    const float* eb_n2g  = (const float*)d_in[25];
    const float* eb_outb = (const float*)d_in[26];
    const float* eb_outw = (const float*)d_in[27];
    const float* eb_pw   = (const float*)d_in[28];
    const float* eb_pwb  = (const float*)d_in[29];
    const float* hg_b = (const float*)d_in[30];
    const float* hg_w = (const float*)d_in[31];
    const float* hl_b = (const float*)d_in[32];
    const float* hl_w = (const float*)d_in[33];
    const float* mb_dw   = (const float*)d_in[34];
    const float* mb_dwb  = (const float*)d_in[35];
    const float* mb_ff1b = (const float*)d_in[36];
    const float* mb_ff1w = (const float*)d_in[37];
    const float* mb_ff2b = (const float*)d_in[38];
    const float* mb_ff2w = (const float*)d_in[39];
    const float* mb_inb  = (const float*)d_in[40];
    const float* mb_inw  = (const float*)d_in[41];
    const float* mb_lnb  = (const float*)d_in[42];
    const float* mb_lng  = (const float*)d_in[43];
    const float* mb_n1b  = (const float*)d_in[44];
    const float* mb_n1g  = (const float*)d_in[45];
    const float* mb_n2b  = (const float*)d_in[46];
    const float* mb_n2g  = (const float*)d_in[47];
    const float* mb_outb = (const float*)d_in[48];
    const float* mb_outw = (const float*)d_in[49];
    const float* mb_pw   = (const float*)d_in[50];
    const float* mb_pwb  = (const float*)d_in[51];
    const float* p1_w       = (const float*)d_in[52];
    const float* p2_w       = (const float*)d_in[53];
    const float* q_resize_w = (const float*)d_in[54];
    const float* resize_g_w = (const float*)d_in[55];
    const float* word_emb   = (const float*)d_in[56];

    float* out = (float*)d_out;

    // workspace layout (floats)
    float* W = (float*)d_ws;
    long long off = 0;
    float* buf_c   = W + off; off += (long long)BLc * H;
    float* buf_q   = W + off; off += (long long)BLq * H;
    float* tmp1    = W + off; off += (long long)BLc * 512;
    float* tmp2    = W + off; off += (long long)BLc * 512;
    float* scores  = W + off; off += (long long)B * LC * LC;
    float* s0      = W + off; off += (long long)B * LC * LQ;
    float* S1      = W + off; off += (long long)B * LC * LQ;
    float* S2      = W + off; off += (long long)B * LC * LQ;
    float* M0      = W + off; off += (long long)BLc * H;
    float* M1      = W + off; off += (long long)BLc * H;
    float* M2      = W + off; off += (long long)BLc * H;
    float* cmask   = W + off; off += BLc;
    float* qmask   = W + off; off += BLq;
    float* cA      = W + off; off += BLc;
    float* qA      = W + off; off += BLq;

    // f16 weight arena (once per launch; ~4.1 MB, L2-resident)
    _Float16* arena = (_Float16*)(W + off);
    _Float16* h_eb_inw  = cvtW(stream, eb_inw,  arena, 3 * H * H);
    _Float16* h_eb_outw = cvtW(stream, eb_outw, arena, H * H);
    _Float16* h_eb_ff1w = cvtW(stream, eb_ff1w, arena, (long long)FF_DIM * H);
    _Float16* h_eb_ff2w = cvtW(stream, eb_ff2w, arena, (long long)H * FF_DIM);
    _Float16* h_eb_pw   = cvtW(stream, eb_pw,   arena, 4LL * H * H);
    _Float16* h_mb_inw  = cvtW(stream, mb_inw,  arena, 7LL * 3 * H * H);
    _Float16* h_mb_outw = cvtW(stream, mb_outw, arena, 7LL * H * H);
    _Float16* h_mb_ff1w = cvtW(stream, mb_ff1w, arena, 7LL * FF_DIM * H);
    _Float16* h_mb_ff2w = cvtW(stream, mb_ff2w, arena, 7LL * H * FF_DIM);
    _Float16* h_mb_pw   = cvtW(stream, mb_pw,   arena, 7LL * 2 * H * H);
    _Float16* h_hl_w    = cvtW(stream, hl_w,    arena, 2LL * H * H);
    _Float16* h_hg_w    = cvtW(stream, hg_w,    arena, 2LL * H * H);
    _Float16* h_crs     = cvtW(stream, c_resize_w, arena, (long long)H * 200);
    _Float16* h_qrs     = cvtW(stream, q_resize_w, arena, (long long)H * 200);
    _Float16* h_grs     = cvtW(stream, resize_g_w, arena, (long long)H * 512);

    mask_kernel<<<(BLc + 255) / 256, 256, 0, stream>>>(c_word, cmask, BLc);
    mask_kernel<<<(BLq + 255) / 256, 256, 0, stream>>>(q_word, qmask, BLq);

    BlockParams embp = { eb_dw, eb_dwb, eb_pwb, eb_lng, eb_lnb, h_eb_pw, 4, 7,
                         h_eb_inw, h_eb_outw, h_eb_ff1w, h_eb_ff2w,
                         eb_inb, eb_outb, eb_ff1b, eb_ff2b,
                         eb_n1g, eb_n1b, eb_n2g, eb_n2b };

    // query path
    charconv_kernel<<<BLq, 128, 0, stream>>>(q_char, char_emb, char_conv_w, char_conv_b, tmp1, 200);
    wordgather_kernel<<<BLq, 128, 0, stream>>>(q_word, word_emb, tmp1, 200);
    gemmW(stream, tmp1, h_qrs, nullptr, nullptr, buf_q, BLq, H, 200, H, 0);
    for (int i = 0; i < 2; ++i) {
        gemmW(stream, buf_q, h_hl_w + (long long)i * H * H, hl_b + i * H, nullptr, tmp1, BLq, H, H, H, 1);
        gemmW(stream, buf_q, h_hg_w + (long long)i * H * H, hg_b + i * H, nullptr, tmp2, BLq, H, H, H, 2);
        highway_combine_kernel<<<((long long)BLq * H + 255) / 256, 256, 0, stream>>>(
            buf_q, tmp1, tmp2, (long long)BLq * H);
    }
    encoder_block(stream, buf_q, B, LQ, qmask, embp, tmp1, tmp2, scores);

    // context path
    charconv_kernel<<<BLc, 128, 0, stream>>>(c_char, char_emb, char_conv_w, char_conv_b, tmp1, 200);
    wordgather_kernel<<<BLc, 128, 0, stream>>>(c_word, word_emb, tmp1, 200);
    gemmW(stream, tmp1, h_crs, nullptr, nullptr, buf_c, BLc, H, 200, H, 0);
    for (int i = 0; i < 2; ++i) {
        gemmW(stream, buf_c, h_hl_w + (long long)i * H * H, hl_b + i * H, nullptr, tmp1, BLc, H, H, H, 1);
        gemmW(stream, buf_c, h_hg_w + (long long)i * H * H, hg_b + i * H, nullptr, tmp2, BLc, H, H, H, 2);
        highway_combine_kernel<<<((long long)BLc * H + 255) / 256, 256, 0, stream>>>(
            buf_c, tmp1, tmp2, (long long)BLc * H);
    }
    encoder_block(stream, buf_c, B, LC, cmask, embp, tmp1, tmp2, scores);

    // attention flow
    float* cs  = tmp2;
    float* c2q = tmp2 + (long long)BLc * H;
    float* q2c = tmp2 + (long long)BLc * H * 2;
    colscale_kernel<<<BLc, H, 0, stream>>>(buf_c, att_cq, cs, H);
    gemm(stream, cs, buf_q, nullptr, nullptr, s0, LC, LQ, H, H, H, LQ,
         (long long)LC * H, (long long)LQ * H, (long long)LC * LQ, 0, B, 1, 0);
    rowdot_kernel<<<BLc, 32, 0, stream>>>(buf_c, att_c, cA, H);
    rowdot_kernel<<<BLq, 32, 0, stream>>>(buf_q, att_q, qA, H);
    builds_kernel<<<dim3(LC, B), 64, 0, stream>>>(s0, cA, qA, att_bias, LC, LQ);
    softmax_kernel<<<dim3(LC, B), 32, 0, stream>>>(s0, S1, qmask, LQ,
        (long long)LC * LQ, (long long)LQ, 1, (long long)LQ, 1.f);
    softmax_kernel<<<dim3(LQ, B), 32, 0, stream>>>(s0, S2, cmask, LC,
        (long long)LC * LQ, 1LL, LQ, (long long)LC, 1.f);
    gemm(stream, S1, buf_q, nullptr, nullptr, c2q, LC, H, LQ, LQ, H, H,
         (long long)LC * LQ, (long long)LQ * H, (long long)LC * H, 0, B, 0, 0);
    gemm(stream, S1, S2, nullptr, nullptr, scores, LC, LC, LQ, LQ, LQ, LC,
         (long long)LC * LQ, (long long)LC * LQ, (long long)LC * LC, 0, B, 1, 0);
    gemm(stream, scores, buf_c, nullptr, nullptr, q2c, LC, H, LC, LC, H, H,
         (long long)LC * LC, (long long)LC * H, (long long)LC * H, 0, B, 0, 0);
    gconcat_kernel<<<BLc, 128, 0, stream>>>(buf_c, c2q, q2c, tmp1);
    gemmW(stream, tmp1, h_grs, nullptr, nullptr, M0, BLc, H, 512, H, 0);

    // model blocks: 7 stacked blocks, run 3 times
    float* Ms[3] = { M0, M1, M2 };
    for (int run = 0; run < 3; ++run) {
        float* X = Ms[run];
        if (run > 0)
            hipMemcpyAsync(X, Ms[run - 1], (size_t)BLc * H * sizeof(float),
                           hipMemcpyDeviceToDevice, stream);
        for (int j = 0; j < 7; ++j) {
            BlockParams p = {
                mb_dw  + (long long)j * 2 * H * 5,
                mb_dwb + (long long)j * 2 * H,
                mb_pwb + (long long)j * 2 * H,
                mb_lng + (long long)j * 2 * H,
                mb_lnb + (long long)j * 2 * H,
                h_mb_pw + (long long)j * 2 * H * H,
                2, 5,
                h_mb_inw  + (long long)j * 3 * H * H,
                h_mb_outw + (long long)j * H * H,
                h_mb_ff1w + (long long)j * FF_DIM * H,
                h_mb_ff2w + (long long)j * H * FF_DIM,
                mb_inb  + (long long)j * 3 * H,
                mb_outb + (long long)j * H,
                mb_ff1b + (long long)j * FF_DIM,
                mb_ff2b + (long long)j * H,
                mb_n1g + (long long)j * H, mb_n1b + (long long)j * H,
                mb_n2g + (long long)j * H, mb_n2b + (long long)j * H };
            encoder_block(stream, X, B, LC, cmask, p, tmp1, tmp2, scores);
        }
    }

    pout_kernel<<<BLc, 32, 0, stream>>>(M0, M1, p1_w, cmask, out);
    pout_kernel<<<BLc, 32, 0, stream>>>(M0, M2, p2_w, cmask, out + BLc);
}
